// SNN_Model_4690104287597
// MI455X (gfx1250) — matrix operand order repaired and checked by reading, not compiled, not run on hardware
//
#include <hip/hip_runtime.h>
#include <math.h>

// ---------------------------------------------------------------- constants
#define BATCH   16384
#define D_IN    784
#define D_PAD   800          // 784 padded to multiple of 32 for WMMA K
#define H1      512
#define H2      10
#define WINS    20
#define THRESH  0.3f
#define DECAYF  0.3f
#define W_DECAY 0.95f
#define TAU_W   40.0f
#define KSLICES 4            // K-split for the Hebbian GEMM (K=16384 -> 4x4096)
#define KSL     (BATCH / KSLICES)

typedef __attribute__((ext_vector_type(16))) __bf16 v16bf;
typedef __attribute__((ext_vector_type(8)))  float  v8f;

// ---------------------------------------------------------------- WMMA helpers
__device__ __forceinline__ v8f wmma_bf16(v16bf a, v16bf b, v8f c) {
    // D = A(16x32) * B(32x16) + C ; f32 accumulate
    return __builtin_amdgcn_wmma_f32_16x16x32_bf16(false, a, false, b,
                                                   (short)0, c, false, false);
}

// A fragment from a per-lane base pointer: row = A + m*lda + (lane>>4)*16, then +k0.
// A layout (matches B): elements e=0..15 hold K = k0+e+16h (contiguous)
__device__ __forceinline__ v16bf load_a_frag_p(const __bf16* __restrict__ row) {
    union { v16bf v; uint4 u[2]; } f;
    f.u[0] = *(const uint4*)(row);
    f.u[1] = *(const uint4*)(row + 8);
    return f.v;
}

// B fragment from a per-lane base pointer: p = Bt + n*ldb + (lane>>4)*16, then +k0.
// 32x16 bf16 B layout used here (matches A): lane n=l&15, h=l>>4; elements e=0..15 hold K = k0+e+16h (contiguous)
__device__ __forceinline__ v16bf load_b_frag_p(const __bf16* __restrict__ p) {
    union { v16bf v; uint4 u[2]; } f;
    f.u[0] = *(const uint4*)(p);
    f.u[1] = *(const uint4*)(p + 8);
    return f.v;
}

// ---------------------------------------------------------------- prep kernels
__global__ void prep_x_kernel(const float* __restrict__ x, const float* __restrict__ beta1,
                              __bf16* __restrict__ xbf, __bf16* __restrict__ XT1) {
    size_t i = (size_t)blockIdx.x * blockDim.x + threadIdx.x;
    if (i >= (size_t)BATCH * D_PAD) return;
    const int b  = (int)(i / D_PAD);
    const int dd = (int)(i % D_PAD);
    float v = (dd < D_IN) ? x[(size_t)b * D_IN + dd] : 0.0f;
    xbf[i] = (__bf16)v;
    if (dd < D_IN)
        XT1[(size_t)dd * BATCH + b] = (__bf16)(v * beta1[dd]);   // (x*beta1)^T
}

__global__ void prep_w1_kernel(const float* __restrict__ W1, __bf16* __restrict__ W1bf) {
    int i = blockIdx.x * blockDim.x + threadIdx.x;
    if (i >= H1 * D_PAD) return;
    const int n = i / D_PAD, k = i % D_PAD;
    W1bf[i] = (__bf16)((k < D_IN) ? W1[n * D_IN + k] : 0.0f);    // W1 is [H1][D] = N-major
}

__global__ void prep_hebb1_kernel(const float* __restrict__ hebb1_in,
                                  float* __restrict__ hebb1f, __bf16* __restrict__ hebb1bf) {
    int i = blockIdx.x * blockDim.x + threadIdx.x;
    if (i >= H1 * D_PAD) return;
    const int h = i / D_PAD, d = i % D_PAD;
    float v = (d < D_IN) ? hebb1_in[d * H1 + h] : 0.0f;
    hebb1bf[i] = (__bf16)v;                                      // N-major [h][d]
    if (d < D_IN) hebb1f[d * H1 + h] = v;
}

__global__ void prep_hebb2_kernel(const float* __restrict__ hebb2_in, float* __restrict__ hebb2f) {
    int i = blockIdx.x * blockDim.x + threadIdx.x;
    if (i < H1 * H2) hebb2f[i] = hebb2_in[i];
}

// ---------------------------------------------------------------- xW1 = x @ W1^T (once)
// wave tile 16x64 (4 accumulators, no spills); block = 8 waves stacked in M -> 128x64 tile
__global__ void gemm_xw1_kernel(const __bf16* __restrict__ A, const __bf16* __restrict__ Bt,
                                float* __restrict__ C) {
    const int lane = threadIdx.x & 31;
    const int wave = threadIdx.x >> 5;
    const int m_base = blockIdx.x * 128 + wave * 16;
    const int n_base = blockIdx.y * 64;
    const int nl = lane & 15, hh = lane >> 4;
    const __bf16* ap = A  + (size_t)(m_base + nl) * D_PAD + hh * 16;
    const __bf16* bp = Bt + (size_t)(n_base + nl) * D_PAD + hh * 16;
    v8f acc[4] = {};
    for (int k0 = 0; k0 < D_PAD; k0 += 32) {
        v16bf a = load_a_frag_p(ap + k0);
#pragma unroll
        for (int j = 0; j < 4; ++j)   // fragment j at constant offset j*16*D_PAD elements
            acc[j] = wmma_bf16(a, load_b_frag_p(bp + j * (16 * D_PAD) + k0), acc[j]);
    }
#pragma unroll
    for (int j = 0; j < 4; ++j)
#pragma unroll
        for (int r = 0; r < 8; ++r)
            C[(size_t)(m_base + r + 8 * hh) * H1 + n_base + j * 16 + nl] = acc[j][r];
}

// ------------------------------------------- per-step layer-1: G1 = x@hebb1, fused update
__global__ void step1_kernel(const __bf16* __restrict__ xbf, const __bf16* __restrict__ hebb1bf,
                             const float* __restrict__ xW1, const float* __restrict__ b1,
                             const float* __restrict__ alpha1p, const float* __restrict__ eta1,
                             float* __restrict__ mem1, float* __restrict__ h1sum,
                             __bf16* __restrict__ post1T, float df) {
    const int lane = threadIdx.x & 31;
    const int wave = threadIdx.x >> 5;
    const int m_base = blockIdx.x * 128 + wave * 16;
    const int n_base = blockIdx.y * 64;
    const int nl = lane & 15, hh = lane >> 4;
    const __bf16* ap = xbf     + (size_t)(m_base + nl) * D_PAD + hh * 16;
    const __bf16* bp = hebb1bf + (size_t)(n_base + nl) * D_PAD + hh * 16;
    v8f acc[4] = {};
    for (int k0 = 0; k0 < D_PAD; k0 += 32) {
        v16bf a = load_a_frag_p(ap + k0);
#pragma unroll
        for (int j = 0; j < 4; ++j)
            acc[j] = wmma_bf16(a, load_b_frag_p(bp + j * (16 * D_PAD) + k0), acc[j]);
    }
    const float a1 = alpha1p[0];
#pragma unroll
    for (int j = 0; j < 4; ++j) {
        const int n   = n_base + j * 16 + nl;
        const float bv = b1[n];
        const float ev = eta1[n];
        union { uint4 q; __bf16 h[8]; } pk;
#pragma unroll
        for (int r = 0; r < 8; ++r) {
            const int m = m_base + r + 8 * hh;
            const size_t idx = (size_t)m * H1 + n;
            // state = df*(x@W1^T) + b1 + alpha1*df*(x@hebb1)
            const float state = df * xW1[idx] + bv + a1 * df * acc[j][r];
            const float mo = mem1[idx];
            const float sp = (mo > THRESH) ? 1.0f : 0.0f;   // previous spike
            const float mn = (mo - sp * THRESH) * DECAYF + state;
            mem1[idx] = mn;
            h1sum[idx] += (mn > THRESH) ? 1.0f : 0.0f;      // new spike
            pk.h[r] = (__bf16)tanhf(mn * (1.0f / THRESH) - ev);
        }
        // post^T [H1][BATCH], bf16: 8 consecutive b per lane -> one 16B store
        *(uint4*)(post1T + (size_t)n * BATCH + m_base + 8 * hh) = pk.q;
    }
}

// ------------------------------------------- per-step hebb1 GEMM, K split into 4 slices
// partial[s][d][h] = sum_{k in slice s} XT1[d][k] * post1T[h][k]
__global__ void hebb1_gemm_kernel(const __bf16* __restrict__ XT1,
                                  const __bf16* __restrict__ post1T,
                                  float* __restrict__ partial) {
    const int lane = threadIdx.x & 31;
    const int wave = threadIdx.x >> 5;
    const int tile = blockIdx.x * 8 + wave;        // 49*32*KSLICES = 6272 tiles exactly
    const int s    = tile / (49 * 32);             // K slice
    const int rem  = tile % (49 * 32);
    const int d_base = (rem >> 5) * 16;            // M tile (784 rows)
    const int n_base = (rem & 31) * 16;            // N tile (512 cols)
    const int nl = lane & 15, hh = lane >> 4;
    const __bf16* ap = XT1    + (size_t)(d_base + nl) * BATCH + hh * 16;
    const __bf16* bp = post1T + (size_t)(n_base + nl) * BATCH + hh * 16;
    v8f acc = {};
    const int kend = (s + 1) * KSL;
    for (int k0 = s * KSL; k0 < kend; k0 += 32)
        acc = wmma_bf16(load_a_frag_p(ap + k0), load_b_frag_p(bp + k0), acc);
    const int n = n_base + nl;
    float* pslice = partial + (size_t)s * D_IN * H1;
#pragma unroll
    for (int r = 0; r < 8; ++r) {
        const int d = d_base + r + 8 * hh;
        pslice[(size_t)d * H1 + n] = acc[r];
    }
}

// reduce the 4 K-slice partials, apply decay+clip, write f32 + N-major bf16 copies
__global__ void hebb1_fixup_kernel(const float* __restrict__ partial,
                                   float* __restrict__ hebb1f, __bf16* __restrict__ hebb1bf,
                                   float df) {
    int i = blockIdx.x * blockDim.x + threadIdx.x;  // over 784*512
    if (i >= D_IN * H1) return;
    const int d = i / H1, h = i % H1;
    float ssum = 0.0f;
#pragma unroll
    for (int s = 0; s < KSLICES; ++s)
        ssum += partial[(size_t)s * D_IN * H1 + i];
    float v = W_DECAY * hebb1f[i] + df * (1.0f / (float)BATCH) * ssum;
    v = fminf(fmaxf(v, -4.0f), 4.0f);
    hebb1f[i] = v;
    hebb1bf[(size_t)h * D_PAD + d] = (__bf16)v;     // N-major bf16 for next step's GEMM
}

// ------------------------------------------- per-step layer-2 forward (tiny: H2=10)
__global__ void layer2_kernel(const float* __restrict__ mem1, const float* __restrict__ W2,
                              const float* __restrict__ b2, const float* __restrict__ alpha2p,
                              const float* __restrict__ eta2, const float* __restrict__ hebb2f,
                              float* __restrict__ mem2, float* __restrict__ h2sum,
                              float* __restrict__ post2, float df) {
    int idx = blockIdx.x * blockDim.x + threadIdx.x;
    if (idx >= BATCH * H2) return;
    const int b = idx / H2, j = idx % H2;
    const float a2 = alpha2p[0];
    const float* m1row = mem1 + (size_t)b * H1;
    const float* w2row = W2 + j * H1;
    float s = 0.0f;
    for (int k = 0; k < H1; ++k) {
        const float sp = (m1row[k] > THRESH) ? 1.0f : 0.0f;   // h1s (new spike)
        s += sp * (w2row[k] + a2 * hebb2f[k * H2 + j]);
    }
    const float state = df * s + b2[j];
    const float mo = mem2[idx];
    const float so = (mo > THRESH) ? 1.0f : 0.0f;
    const float mn = (mo - so * THRESH) * DECAYF + state;
    mem2[idx] = mn;
    h2sum[idx] += (mn > THRESH) ? 1.0f : 0.0f;
    post2[idx] = tanhf(mn * (1.0f / THRESH) - eta2[j]);
}

// ------------------------------------------- per-step hebb2 update (512x10, K=16384 reduce)
__global__ void hebb2_update_kernel(const float* __restrict__ mem1, const float* __restrict__ post2,
                                    const float* __restrict__ beta2, float* __restrict__ hebb2f,
                                    float df) {
    const int k = blockIdx.x;        // 0..511
    const int t = threadIdx.x;       // 256 threads
    float acc[H2];
#pragma unroll
    for (int j = 0; j < H2; ++j) acc[j] = 0.0f;
    for (int b = t; b < BATCH; b += 256) {
        if (mem1[(size_t)b * H1 + k] > THRESH) {   // h1s == 1
            const float* p2 = post2 + (size_t)b * H2;
#pragma unroll
            for (int j = 0; j < H2; ++j) acc[j] += p2[j];
        }
    }
    __shared__ float red[256];
    const float scale = df * beta2[k] * (1.0f / (float)BATCH);
    for (int j = 0; j < H2; ++j) {
        red[t] = acc[j];
        __syncthreads();
        for (int s = 128; s > 0; s >>= 1) {
            if (t < s) red[t] += red[t + s];
            __syncthreads();
        }
        if (t == 0) {
            float v = W_DECAY * hebb2f[k * H2 + j] + scale * red[0];
            hebb2f[k * H2 + j] = fminf(fmaxf(v, -4.0f), 4.0f);
        }
        __syncthreads();
    }
}

// ------------------------------------------- finalize: outs + eta copies
__global__ void finalize_kernel(const float* __restrict__ mem2, float* __restrict__ outs,
                                const float* __restrict__ eta1, const float* __restrict__ eta2,
                                float* __restrict__ oeta1, float* __restrict__ oeta2) {
    int i = blockIdx.x * blockDim.x + threadIdx.x;
    if (i < BATCH * H2) outs[i] = fminf(mem2[i] * (1.0f / THRESH), 1.1f);
    if (i < H1) oeta1[i] = eta1[i];
    if (i < H2) oeta2[i] = eta2[i];
}

// ---------------------------------------------------------------- host launcher
extern "C" void kernel_launch(void* const* d_in, const int* in_sizes, int n_in,
                              void* d_out, int out_size, void* d_ws, size_t ws_size,
                              hipStream_t stream) {
    const float* x        = (const float*)d_in[0];
    const float* hebb1_in = (const float*)d_in[1];
    const float* hebb2_in = (const float*)d_in[2];
    const float* W1       = (const float*)d_in[3];
    const float* b1       = (const float*)d_in[4];
    const float* W2       = (const float*)d_in[5];
    const float* b2       = (const float*)d_in[6];
    const float* alpha1   = (const float*)d_in[7];
    const float* alpha2   = (const float*)d_in[8];
    const float* beta1    = (const float*)d_in[9];
    const float* beta2    = (const float*)d_in[10];
    const float* eta1     = (const float*)d_in[11];
    const float* eta2     = (const float*)d_in[12];

    // d_out layout (return order, f32)
    float* out   = (float*)d_out;
    float* outs  = out;                                   // [B, H2]
    float* h1sum = out + (size_t)BATCH * H2;              // [B, H1]
    float* h2sum = h1sum + (size_t)BATCH * H1;            // [B, H2]
    float* hb1   = h2sum + (size_t)BATCH * H2;            // [D, H1]  (working buffer too)
    float* hb2   = hb1 + (size_t)D_IN * H1;               // [H1, H2] (working buffer too)
    float* oeta1 = hb2 + (size_t)H1 * H2;                 // [H1]
    float* oeta2 = oeta1 + H1;                            // [H2]

    // workspace carve-up (256B aligned slices)
    char* w = (char*)d_ws;
    size_t used = 0;
    auto alloc = [&](size_t bytes) {
        char* p = w + used;
        used += (bytes + 255) & ~(size_t)255;
        return p;
    };
    __bf16* xbf     = (__bf16*)alloc((size_t)BATCH * D_PAD * 2);  // x padded, bf16
    __bf16* XT1     = (__bf16*)alloc((size_t)D_IN * BATCH * 2);   // (x*beta1)^T bf16
    __bf16* W1bf    = (__bf16*)alloc((size_t)H1 * D_PAD * 2);     // W1 N-major bf16
    __bf16* hebb1bf = (__bf16*)alloc((size_t)H1 * D_PAD * 2);     // hebb1 N-major bf16
    float*  xW1     = (float*)alloc((size_t)BATCH * H1 * 4);      // x @ W1^T
    float*  mem1    = (float*)alloc((size_t)BATCH * H1 * 4);
    __bf16* post1T  = (__bf16*)alloc((size_t)H1 * BATCH * 2);     // post^T bf16
    float*  partial = (float*)alloc((size_t)KSLICES * D_IN * H1 * 4);  // hebb1 K-slices
    float*  mem2    = (float*)alloc((size_t)BATCH * H2 * 4);
    float*  post2   = (float*)alloc((size_t)BATCH * H2 * 4);
    if (used > ws_size) return;   // workspace too small; bail deterministically

    // ---- init state (every call: deterministic)
    hipMemsetAsync(mem1,  0, (size_t)BATCH * H1 * 4, stream);
    hipMemsetAsync(mem2,  0, (size_t)BATCH * H2 * 4, stream);
    hipMemsetAsync(h1sum, 0, (size_t)BATCH * H1 * 4, stream);
    hipMemsetAsync(h2sum, 0, (size_t)BATCH * H2 * 4, stream);

    const int TB = 256;
    prep_x_kernel<<<(int)(((size_t)BATCH * D_PAD + TB - 1) / TB), TB, 0, stream>>>(x, beta1, xbf, XT1);
    prep_w1_kernel<<<(H1 * D_PAD + TB - 1) / TB, TB, 0, stream>>>(W1, W1bf);
    prep_hebb1_kernel<<<(H1 * D_PAD + TB - 1) / TB, TB, 0, stream>>>(hebb1_in, hb1, hebb1bf);
    prep_hebb2_kernel<<<(H1 * H2 + TB - 1) / TB, TB, 0, stream>>>(hebb2_in, hb2);

    // hoisted GEMM: xW1 = x @ W1^T  (time factor df applied per-step)
    gemm_xw1_kernel<<<dim3(BATCH / 128, H1 / 64), TB, 0, stream>>>(xbf, W1bf, xW1);

    // ---- time loop
    for (int t = 0; t < WINS; ++t) {
        const float df = expf(-(float)t / TAU_W);
        step1_kernel<<<dim3(BATCH / 128, H1 / 64), TB, 0, stream>>>(
            xbf, hebb1bf, xW1, b1, alpha1, eta1, mem1, h1sum, post1T, df);
        hebb1_gemm_kernel<<<(49 * 32 * KSLICES) / 8, TB, 0, stream>>>(XT1, post1T, partial);
        hebb1_fixup_kernel<<<(D_IN * H1 + TB - 1) / TB, TB, 0, stream>>>(partial, hb1, hebb1bf, df);
        layer2_kernel<<<(BATCH * H2 + TB - 1) / TB, TB, 0, stream>>>(
            mem1, W2, b2, alpha2, eta2, hb2, mem2, h2sum, post2, df);
        hebb2_update_kernel<<<H1, TB, 0, stream>>>(mem1, post2, beta2, hb2, df);
    }

    finalize_kernel<<<(BATCH * H2 + TB - 1) / TB, TB, 0, stream>>>(
        mem2, outs, eta1, eta2, oeta1, oeta2);
}
